// HierarchicalFeaturesExtractor_12618613916384
// MI455X (gfx1250) — matrix operand order repaired and checked by reading, not compiled, not run on hardware
//
#include <hip/hip_runtime.h>
#include <hip/hip_bf16.h>

// ---------------------------------------------------------------------------
// HierarchicalFeaturesExtractor, fused for CDNA5 (gfx1250, wave32, WMMA).
//
// B=64, S=512, E=256, D=512.  out[b,s,d]:
//   s >= 256 : b1 + W1 . x[b,s]                                   (K=256)
//   2<=s<256 : bias_lo + sum_{t=-2..3} Weff_t . x[b,s+t]          (K=1536)
//   s in 0,1 : exact fp32 fixup (truncated windows)
// Weff folds conv1 + the windowed sums of the 2/3/4-gram branches into one
// 6-tap conv.  fp16 WMMA (v_wmma_f32_16x16x32_f16) with fp32 accumulation.
// ---------------------------------------------------------------------------

typedef _Float16 hv4  __attribute__((ext_vector_type(4)));
typedef _Float16 hv8  __attribute__((ext_vector_type(8)));
typedef _Float16 hv16 __attribute__((ext_vector_type(16)));
typedef float    f32x8 __attribute__((ext_vector_type(8)));

#define BB 64
#define SS 512
#define EE 256
#define DD 512
#define LDS_STRIDE 264   // 256 halves + 8 pad -> 132 dwords/row, bank-conflict free

__device__ __forceinline__ hv16 cat8(hv8 a, hv8 b) {
    return __builtin_shufflevector(a, b, 0,1,2,3,4,5,6,7,8,9,10,11,12,13,14,15);
}

// ---------------------------------------------------------------------------
// Prep: build Weff[6][D][E] (f16), W1h[D][E] (f16), bias_lo/hi[D] (f32).
// Weff tap index t=0..5 corresponds to x-offset t-2.
// ---------------------------------------------------------------------------
__global__ __launch_bounds__(256) void hfe_prep(
    const float* __restrict__ W1, const float* __restrict__ b1,
    const float* __restrict__ W2, const float* __restrict__ b2,
    const float* __restrict__ W3, const float* __restrict__ b3,
    const float* __restrict__ W4, const float* __restrict__ b4,
    _Float16* __restrict__ Weff, _Float16* __restrict__ W1h,
    float* __restrict__ bias_lo, float* __restrict__ bias_hi)
{
    const int idx = blockIdx.x * 256 + threadIdx.x;
    if (idx < DD * EE) {
        const int d = idx >> 8;
        const int e = idx & 255;
        const float  w1  = W1[(size_t)d * EE + e];               // [D,E,1]
        const float* w2  = W2 + ((size_t)d * EE + e) * 2;        // [D,E,2]
        const float* w3  = W3 + ((size_t)d * EE + e) * 3;        // [D,E,3]
        const float* w4  = W4 + ((size_t)d * EE + e) * 4;        // [D,E,4]
        const float w20 = w2[0], w21 = w2[1];
        const float w30 = w3[0], w31 = w3[1], w32 = w3[2];
        const float w40 = w4[0], w41 = w4[1], w42 = w4[2], w43 = w4[3];
        float tv[6];
        tv[0] = w40;                                             // t = -2
        tv[1] = w30 + w40 + w41;                                 // t = -1
        tv[2] = w1 + w20 + w30 + w31 + w40 + w41 + w42;          // t =  0
        tv[3] = w21 + w31 + w32 + w41 + w42 + w43;               // t = +1
        tv[4] = w32 + w42 + w43;                                 // t = +2
        tv[5] = w43;                                             // t = +3
        #pragma unroll
        for (int t = 0; t < 6; ++t)
            Weff[(size_t)t * DD * EE + idx] = (_Float16)tv[t];
        W1h[idx] = (_Float16)w1;
    }
    if (idx < DD) {
        bias_lo[idx] = b1[idx] + b2[idx] + 2.0f * b3[idx] + 3.0f * b4[idx];
        bias_hi[idx] = b1[idx];
    }
}

// ---------------------------------------------------------------------------
// Main GEMM: one workgroup (256 thr = 8 wave32) computes a 16(s) x 512(d)
// output tile; K = NTAPS*256 via v_wmma_f32_16x16x32_f16.
//   NROWS feature rows staged in LDS as f16 (row r <-> s = s_base + r + TOFF).
//   Wave w owns d-columns [64w, 64w+64): 4 accumulators of 16x16 f32.
// ---------------------------------------------------------------------------
template <int NTAPS, int TOFF, int NROWS>
__global__ __launch_bounds__(256) void hfe_gemm(
    const float* __restrict__ X,          // [B,S,E] fp32
    const _Float16* __restrict__ Wt,      // [NTAPS,D,E] f16
    const float* __restrict__ bias,       // [D]
    float* __restrict__ out,              // [B,S,D]
    int region_base)
{
    __shared__ _Float16 sA[NROWS * LDS_STRIDE];

    const int tid    = threadIdx.x;
    const int b      = blockIdx.x >> 4;
    const int s_base = region_base + ((blockIdx.x & 15) << 4);

    // ---- stage A rows into LDS (fp32 -> f16 on the fly, b128 global loads)
    for (int idx4 = tid; idx4 < NROWS * (EE / 4); idx4 += 256) {
        const int r  = idx4 >> 6;          // row 0..NROWS-1
        const int e4 = (idx4 & 63) << 2;   // e, multiple of 4
        int s = s_base + r + TOFF;
        s = (s < 0) ? 0 : s;               // clamp; s=0,1 outputs fixed later
        const float4 v = *(const float4*)(X + ((size_t)(b * SS + s)) * EE + e4);
        hv4 h = { (_Float16)v.x, (_Float16)v.y, (_Float16)v.z, (_Float16)v.w };
        *(hv4*)(&sA[r * LDS_STRIDE + e4]) = h;
    }
    __syncthreads();

    const int wave = tid >> 5;
    const int lane = tid & 31;
    const int lm   = lane & 15;    // column / row-in-half selector
    const int lc   = lane >> 4;    // lane half (K split)
    const int n0   = wave << 6;    // 64 d-columns per wave

    f32x8 acc[4] = {};

    #pragma unroll
    for (int t = 0; t < NTAPS; ++t) {
        const int ar = lm + t;                               // LDS row for A[m=lm]
        const _Float16* wrow = Wt + (size_t)t * DD * EE;
        #pragma unroll
        for (int e0 = 0; e0 < EE; e0 += 32) {
            // A fragment (ISA 7.12.2): lane half lc holds K = lc*8+0..7 and 16+lc*8+0..7
            const _Float16* ap = &sA[ar * LDS_STRIDE + e0];
            hv8 a0 = *(const hv8*)(ap + lc * 8);
            hv8 a1 = *(const hv8*)(ap + 16 + lc * 8);
            hv16 aF = cat8(a0, a1);
            #pragma unroll
            for (int i = 0; i < 4; ++i) {
                // B fragment: column n = lane%16, K = lc*8+0..7 and 16+lc*8+0..7 (same as A)
                const int n = n0 + i * 16 + lm;
                const _Float16* bp = wrow + (size_t)n * EE + e0 + lc * 8;
                hv8 b0 = *(const hv8*)(bp);
                hv8 b1 = *(const hv8*)(bp + 16);
                hv16 bF = cat8(b0, b1);
                acc[i] = __builtin_amdgcn_wmma_f32_16x16x32_f16(
                    false, aF, false, bF, (short)0, acc[i], false, false);
            }
        }
    }

    // ---- epilogue: C/D layout -> row m = vgpr + 8*lc, col n = lane%16
    #pragma unroll
    for (int i = 0; i < 4; ++i) {
        const int n  = n0 + i * 16 + lm;
        const float bv = bias[n];
        #pragma unroll
        for (int r = 0; r < 8; ++r) {
            const int m = r + (lc << 3);
            out[((size_t)(b * SS + s_base + m)) * DD + n] = acc[i][r] + bv;
        }
    }
}

// ---------------------------------------------------------------------------
// Exact fp32 fixup for s = 0, 1 (truncated windows). 64*2*512 elements.
// ---------------------------------------------------------------------------
__global__ __launch_bounds__(256) void hfe_fix_edges(
    const float* __restrict__ X,
    const float* __restrict__ W1, const float* __restrict__ b1,
    const float* __restrict__ W2, const float* __restrict__ b2,
    const float* __restrict__ W3, const float* __restrict__ b3,
    const float* __restrict__ W4, const float* __restrict__ b4,
    float* __restrict__ out)
{
    const int idx = blockIdx.x * 256 + threadIdx.x;
    if (idx >= BB * 2 * DD) return;
    const int d = idx & 511;
    const int s = (idx >> 9) & 1;
    const int b = idx >> 10;
    const float* x = X + (size_t)b * SS * EE;

    float acc = b1[d];
    { // unigram conv at s
        const float* w = W1 + (size_t)d * EE;
        for (int e = 0; e < EE; ++e) acc += w[e] * x[s * EE + e];
    }
    { // 2-gram branch, window {s}
        float f = b2[d];
        const float* w = W2 + (size_t)d * EE * 2;
        for (int k = 0; k < 2; ++k)
            for (int e = 0; e < EE; ++e) f += w[e * 2 + k] * x[(s + k) * EE + e];
        acc += f;
    }
    { // 3-gram branch, window p = max(0,s-1)..s
        const float* w = W3 + (size_t)d * EE * 3;
        for (int p = (s - 1 < 0 ? 0 : s - 1); p <= s; ++p) {
            float f = b3[d];
            for (int k = 0; k < 3; ++k)
                for (int e = 0; e < EE; ++e) f += w[e * 3 + k] * x[(p + k) * EE + e];
            acc += f;
        }
    }
    { // 4-gram branch, window p = max(0,s-2)..s
        const float* w = W4 + (size_t)d * EE * 4;
        for (int p = (s - 2 < 0 ? 0 : s - 2); p <= s; ++p) {
            float f = b4[d];
            for (int k = 0; k < 4; ++k)
                for (int e = 0; e < EE; ++e) f += w[e * 4 + k] * x[(p + k) * EE + e];
            acc += f;
        }
    }
    out[((size_t)(b * SS + s)) * DD + d] = acc;
}

// ---------------------------------------------------------------------------
extern "C" void kernel_launch(void* const* d_in, const int* in_sizes, int n_in,
                              void* d_out, int out_size, void* d_ws, size_t ws_size,
                              hipStream_t stream) {
    (void)in_sizes; (void)n_in; (void)out_size; (void)ws_size;
    const float* X  = (const float*)d_in[0];
    const float* W1 = (const float*)d_in[1];
    const float* b1 = (const float*)d_in[2];
    const float* W2 = (const float*)d_in[3];
    const float* b2 = (const float*)d_in[4];
    const float* W3 = (const float*)d_in[5];
    const float* b3 = (const float*)d_in[6];
    const float* W4 = (const float*)d_in[7];
    const float* b4 = (const float*)d_in[8];
    float* out = (float*)d_out;

    // workspace carve: Weff (6*D*E f16) | W1h (D*E f16) | bias_lo | bias_hi
    _Float16* Weff    = (_Float16*)d_ws;
    _Float16* W1h     = Weff + (size_t)6 * DD * EE;
    float*    bias_lo = (float*)(W1h + (size_t)DD * EE);
    float*    bias_hi = bias_lo + DD;

    hfe_prep<<<dim3((DD * EE + 255) / 256), dim3(256), 0, stream>>>(
        W1, b1, W2, b2, W3, b3, W4, b4, Weff, W1h, bias_lo, bias_hi);

    // lo region: s in [0,256), 6-tap fused conv, K = 1536
    hfe_gemm<6, -2, 21><<<dim3(BB * 16), dim3(256), 0, stream>>>(
        X, Weff, bias_lo, out, 0);

    // hi region: s in [256,512), unigram conv only, K = 256
    hfe_gemm<1, 0, 16><<<dim3(BB * 16), dim3(256), 0, stream>>>(
        X, W1h, bias_hi, out, 256);

    // exact fixup for truncated windows at s = 0, 1
    hfe_fix_edges<<<dim3((BB * 2 * DD + 255) / 256), dim3(256), 0, stream>>>(
        X, W1, b1, W2, b2, W3, b3, W4, b4, out);
}
